// FusionModel_26620207301074
// MI455X (gfx1250) — matrix operand order repaired and checked by reading, not compiled, not run on hardware
//
#include <hip/hip_runtime.h>

// ---------------------------------------------------------------------------
// Fused MoE for MI455X (gfx1250, wave32, WMMA 16x16x32 f16, f32 accum)
// B=131072 D=64 H=512 T=8 E=4 HG=256
// ---------------------------------------------------------------------------

#define BTOK  131072
#define DDIM  64
#define HDIM  512
#define TDIM  8
#define EEXP  4
#define HGDIM 256

typedef _Float16 v16h __attribute__((ext_vector_type(16)));
typedef float    v8f  __attribute__((ext_vector_type(8)));

// d_ws layout (ushort elements); each fragment = 512 f16 (32 lanes x 16)
#define FR_WG1 (16 * 2)        // NT=16, KT=2
#define FR_WG2 (1 * 8)         // NT=1,  KT=8   (N padded 4->16)
#define FR_WE1 (4 * 32 * 2)    // E=4, NT=32, KT=2
#define FR_WE2 (4 * 32 * 16)   // E=4, NT=32, KT=16
#define FR_WE3 (4 * 1 * 16)    // E=4, NT=1,  KT=16 (N padded 8->16)
#define OFF_WG1 0
#define OFF_WG2 (OFF_WG1 + FR_WG1 * 512)
#define OFF_WE1 (OFF_WG2 + FR_WG2 * 512)
#define OFF_WE2 (OFF_WE1 + FR_WE1 * 512)
#define OFF_WE3 (OFF_WE2 + FR_WE2 * 512)

__device__ __forceinline__ unsigned short f2h(float a) {
    return __builtin_bit_cast(unsigned short, (_Float16)a);
}
__device__ __forceinline__ unsigned int pk2h(float a, float b) {
    return (unsigned int)f2h(a) | ((unsigned int)f2h(b) << 16);
}

// A-fragment (16x32 f16, M x K): lane m = ln, khalf = hl.
// VGPR0-3 hold K = hl*8 + 0..7 ; VGPR4-7 hold K = 16 + hl*8 + 0..7.
// Row-major LDS tile (pitch in ushorts, +8 padding -> conflict-free).
__device__ __forceinline__ v16h lds_afrag(const unsigned short* base, int pitch,
                                          int m, int kcol, int hl) {
    union { uint4 q[2]; v16h v; } u;
    u.q[0] = *(const uint4*)(base + m * pitch + kcol + hl * 8);
    u.q[1] = *(const uint4*)(base + m * pitch + kcol + 16 + hl * 8);
    return u.v;
}

// B-fragment (32x16 f16, K x N) pre-packed in global: lane stores its 16
// contiguous f16 at frag + lane*16.
__device__ __forceinline__ v16h gl_bfrag(const unsigned short* frag, int lane) {
    union { uint4 q[2]; v16h v; } u;
    const uint4* p = (const uint4*)(frag + lane * 16);
    u.q[0] = p[0];
    u.q[1] = p[1];
    return u.v;
}

__device__ __forceinline__ v8f wmma16(v16h a, v16h b, v8f c) {
    return __builtin_amdgcn_wmma_f32_16x16x32_f16(false, a, false, b,
                                                  (short)0, c, false, false);
}

__device__ __forceinline__ v8f splat8(float b) {
    v8f r;
#pragma unroll
    for (int i = 0; i < 8; i++) r[i] = b;
    return r;
}

// ---------------------------------------------------------------------------
// Prepack: fp32 row-major (K x Nvalid) -> f16 WMMA B-fragments.
// blockIdx.x = (e*NT + nt)*KT + kt, 32 threads (one wave) per fragment.
// lane: n = nt*16 + (lane&15), khalf = lane>>4, element i <-> K = kt*32+khalf*8+i (i<8), kt*32+16+khalf*8+(i-8) (i>=8)
// ---------------------------------------------------------------------------
__global__ __launch_bounds__(32) void prepack_f16(const float* __restrict__ src,
                                                  unsigned short* __restrict__ dst,
                                                  int K, int Nvalid, int KT, int NT) {
    int fid = blockIdx.x;
    int kt = fid % KT;
    int rn = fid / KT;
    int nt = rn % NT;
    int e = rn / NT;
    int lane = threadIdx.x;
    int n = nt * 16 + (lane & 15);
    int khalf = lane >> 4;
    const float* s = src + (size_t)e * K * Nvalid;
    unsigned int u[8];
#pragma unroll
    for (int i = 0; i < 8; i++) {
        int k0 = kt * 32 + khalf * 8 + 2 * i + (i >= 4 ? 8 : 0);
        float a = (n < Nvalid && k0 < K) ? s[(size_t)k0 * Nvalid + n] : 0.f;
        float b = (n < Nvalid && (k0 + 1) < K) ? s[(size_t)(k0 + 1) * Nvalid + n] : 0.f;
        u[i] = pk2h(a, b);
    }
    uint4* d = (uint4*)(dst + (size_t)fid * 512 + lane * 16);
    d[0] = make_uint4(u[0], u[1], u[2], u[3]);
    d[1] = make_uint4(u[4], u[5], u[6], u[7]);
}

// ---------------------------------------------------------------------------
// Fused kernel: 64 tokens / block, 256 threads = 8 waves.
// 4 row-strips of 16 tokens, 2 waves per strip (split output columns).
// ---------------------------------------------------------------------------
__global__ __launch_bounds__(256) void moe_fused(
    const float* __restrict__ xg,
    const unsigned short* __restrict__ wg1p, const unsigned short* __restrict__ wg2p,
    const unsigned short* __restrict__ we1p, const unsigned short* __restrict__ we2p,
    const unsigned short* __restrict__ we3p,
    const float* __restrict__ bg1, const float* __restrict__ bg2,
    const float* __restrict__ be1, const float* __restrict__ be2,
    const float* __restrict__ be3,
    float* __restrict__ outF, float* __restrict__ outW, float* __restrict__ outEO) {
    constexpr int XP = DDIM + 8;   // 72
    constexpr int HP = HDIM + 8;   // 520
    constexpr int GP = HGDIM + 8;  // 264

    __shared__ __align__(16) unsigned short xs[64 * XP];
    __shared__ __align__(16) unsigned short hs[64 * HP];
    __shared__ __align__(16) unsigned short gs[64 * GP];
    __shared__ __align__(16) float g2s[64 * 17];
    __shared__ float wlds[64 * 4];

    const int tid = threadIdx.x;
    const int lane = tid & 31;
    const int wave = tid >> 5;
    const int strip = wave >> 1;   // 0..3
    const int cw = wave & 1;       // column half
    const int mbase = strip * 16;
    const int ln = lane & 15;
    const int hl = lane >> 4;
    const int mrow = mbase + ln;
    const size_t gb0 = (size_t)blockIdx.x * 64;

    // ---- stage x tile (64x64 f32) -> xs (f16, padded) ----
    {
        int r = tid >> 2;
        int c0 = (tid & 3) * 16;
        const float4* src = (const float4*)(xg + (gb0 + r) * DDIM + c0);
        unsigned int u[8];
#pragma unroll
        for (int i = 0; i < 4; i++) {
            float4 f = src[i];
            u[2 * i + 0] = pk2h(f.x, f.y);
            u[2 * i + 1] = pk2h(f.z, f.w);
        }
        uint4* dst = (uint4*)&xs[r * XP + c0];
        dst[0] = make_uint4(u[0], u[1], u[2], u[3]);
        dst[1] = make_uint4(u[4], u[5], u[6], u[7]);
    }
    __syncthreads();

    // x A-fragments for this strip (K=64 -> 2 fragments), reused everywhere
    const v16h xa0 = lds_afrag(xs, XP, mrow, 0, hl);
    const v16h xa1 = lds_afrag(xs, XP, mrow, 32, hl);

    // ---- gate layer 1: relu(x @ Wg1 + bg1) -> gs (64x256 f16) ----
    for (int j = 0; j < 8; j++) {
        int nt = cw * 8 + j;
        int ncol = nt * 16 + ln;
        v8f acc = splat8(bg1[ncol]);
        acc = wmma16(xa0, gl_bfrag(wg1p + (size_t)(nt * 2 + 0) * 512, lane), acc);
        acc = wmma16(xa1, gl_bfrag(wg1p + (size_t)(nt * 2 + 1) * 512, lane), acc);
#pragma unroll
        for (int v = 0; v < 8; v++)
            gs[(mbase + v + 8 * hl) * GP + ncol] = f2h(fmaxf(acc[v], 0.f));
    }
    __syncthreads();

    // ---- gate layer 2 logits (even waves): gs @ Wg2pad + bg2 -> g2s ----
    if (cw == 0) {
        v8f acc = splat8(ln < EEXP ? bg2[ln] : 0.f);
#pragma unroll
        for (int kt = 0; kt < 8; kt++) {
            v16h a = lds_afrag(gs, GP, mrow, kt * 32, hl);
            acc = wmma16(a, gl_bfrag(wg2p + (size_t)kt * 512, lane), acc);
        }
#pragma unroll
        for (int v = 0; v < 8; v++)
            g2s[(mbase + v + 8 * hl) * 17 + ln] = acc[v];
    }
    __syncthreads();

    // ---- softmax over E=4, write w to LDS and d_out ----
    if (tid < 64) {
        int r = tid;
        float l0 = g2s[r * 17 + 0], l1 = g2s[r * 17 + 1];
        float l2 = g2s[r * 17 + 2], l3 = g2s[r * 17 + 3];
        float mx = fmaxf(fmaxf(l0, l1), fmaxf(l2, l3));
        float e0 = __expf(l0 - mx), e1 = __expf(l1 - mx);
        float e2 = __expf(l2 - mx), e3 = __expf(l3 - mx);
        float inv = 1.f / (e0 + e1 + e2 + e3);
        float w0 = e0 * inv, w1 = e1 * inv, w2 = e2 * inv, w3 = e3 * inv;
        wlds[r * 4 + 0] = w0; wlds[r * 4 + 1] = w1;
        wlds[r * 4 + 2] = w2; wlds[r * 4 + 3] = w3;
        float4 wv = make_float4(w0, w1, w2, w3);
        ((float4*)outW)[gb0 + r] = wv;
    }
    __syncthreads();

    v8f facc = splat8(0.f);  // fused accumulator (even waves)

    for (int e = 0; e < EEXP; e++) {
        // ---- GEMM1: h1 = relu(x @ We1[e] + be1[e]) -> hs ----
        for (int j = 0; j < 16; j++) {
            int nt = cw * 16 + j;
            int ncol = nt * 16 + ln;
            v8f acc = splat8(be1[e * HDIM + ncol]);
            acc = wmma16(xa0, gl_bfrag(we1p + (size_t)((e * 32 + nt) * 2 + 0) * 512, lane), acc);
            acc = wmma16(xa1, gl_bfrag(we1p + (size_t)((e * 32 + nt) * 2 + 1) * 512, lane), acc);
#pragma unroll
            for (int v = 0; v < 8; v++)
                hs[(mbase + v + 8 * hl) * HP + ncol] = f2h(fmaxf(acc[v], 0.f));
        }
        __syncthreads();

        // ---- preload full-K A-strip for GEMM2 (16 frags = 128 VGPRs) ----
        v16h ha[16];
#pragma unroll
        for (int kt = 0; kt < 16; kt++)
            ha[kt] = lds_afrag(hs, HP, mrow, kt * 32, hl);
        __syncthreads();  // everyone done reading h1 before overwriting

        // ---- GEMM2: h2 = relu(h1 @ We2[e] + be2[e]) -> hs (in place) ----
        for (int j = 0; j < 16; j++) {
            int nt = cw * 16 + j;
            int ncol = nt * 16 + ln;
            v8f acc = splat8(be2[e * HDIM + ncol]);
#pragma unroll
            for (int kt = 0; kt < 16; kt++)
                acc = wmma16(ha[kt],
                             gl_bfrag(we2p + (size_t)((e * 32 + nt) * 16 + kt) * 512, lane),
                             acc);
#pragma unroll
            for (int v = 0; v < 8; v++)
                hs[(mbase + v + 8 * hl) * HP + ncol] = f2h(fmaxf(acc[v], 0.f));
        }
        __syncthreads();

        // ---- GEMM3 (even waves): eo = h2 @ We3pad[e] + be3[e] ----
        if (cw == 0) {
            v8f acc = splat8(ln < TDIM ? be3[e * TDIM + ln] : 0.f);
#pragma unroll
            for (int kt = 0; kt < 16; kt++) {
                v16h a = lds_afrag(hs, HP, mrow, kt * 32, hl);
                acc = wmma16(a, gl_bfrag(we3p + (size_t)(e * 16 + kt) * 512, lane), acc);
            }
            if (ln < TDIM) {
#pragma unroll
                for (int v = 0; v < 8; v++) {
                    int row = mbase + v + 8 * hl;
                    outEO[((gb0 + row) * EEXP + e) * TDIM + ln] = acc[v];
                }
            }
#pragma unroll
            for (int v = 0; v < 8; v++) {
                int row = mbase + v + 8 * hl;
                facc[v] += wlds[row * 4 + e] * acc[v];
            }
        }
        __syncthreads();  // protect hs before next expert's GEMM1
    }

    // ---- fused output (B x T) ----
    if (cw == 0 && ln < TDIM) {
#pragma unroll
        for (int v = 0; v < 8; v++) {
            int row = mbase + v + 8 * hl;
            outF[(gb0 + row) * TDIM + ln] = facc[v];
        }
    }
}

// ---------------------------------------------------------------------------
extern "C" void kernel_launch(void* const* d_in, const int* in_sizes, int n_in,
                              void* d_out, int out_size, void* d_ws, size_t ws_size,
                              hipStream_t stream) {
    const float* x   = (const float*)d_in[0];
    const float* Wg1 = (const float*)d_in[1];
    const float* bg1 = (const float*)d_in[2];
    const float* Wg2 = (const float*)d_in[3];
    const float* bg2 = (const float*)d_in[4];
    const float* We1 = (const float*)d_in[5];
    const float* be1 = (const float*)d_in[6];
    const float* We2 = (const float*)d_in[7];
    const float* be2 = (const float*)d_in[8];
    const float* We3 = (const float*)d_in[9];
    const float* be3 = (const float*)d_in[10];

    unsigned short* ws = (unsigned short*)d_ws;
    unsigned short* wg1p = ws + OFF_WG1;
    unsigned short* wg2p = ws + OFF_WG2;
    unsigned short* we1p = ws + OFF_WE1;
    unsigned short* we2p = ws + OFF_WE2;
    unsigned short* we3p = ws + OFF_WE3;

    // weight prepack (f32 -> f16 WMMA B-fragments), ~2.5 MB total
    prepack_f16<<<FR_WG1, 32, 0, stream>>>(Wg1, wg1p, DDIM, HGDIM, 2, 16);
    prepack_f16<<<FR_WG2, 32, 0, stream>>>(Wg2, wg2p, HGDIM, EEXP, 8, 1);
    prepack_f16<<<FR_WE1, 32, 0, stream>>>(We1, we1p, DDIM, HDIM, 2, 32);
    prepack_f16<<<FR_WE2, 32, 0, stream>>>(We2, we2p, HDIM, HDIM, 16, 32);
    prepack_f16<<<FR_WE3, 32, 0, stream>>>(We3, we3p, HDIM, TDIM, 16, 1);

    float* outF = (float*)d_out;
    float* outW = outF + (size_t)BTOK * TDIM;
    float* outEO = outW + (size_t)BTOK * EEXP;

    moe_fused<<<BTOK / 64, 256, 0, stream>>>(x, wg1p, wg2p, we1p, we2p, we3p,
                                             bg1, bg2, be1, be2, be3,
                                             outF, outW, outEO);
}
